// BiMambaWrapper_61710090109317
// MI455X (gfx1250) — matrix operand order repaired and checked by reading, not compiled, not run on hardware
//
#include <hip/hip_runtime.h>
#include <hip/hip_bf16.h>

// ---------------- problem constants ----------------
constexpr int D_MODEL = 768;
constexpr int D_INNER = 1536;
constexpr int D_STATE = 16;
constexpr int D_CONV  = 4;
constexpr int DT_RANK = 48;
constexpr int XD      = DT_RANK + 2 * D_STATE;   // 80
constexpr int BATCH   = 2;
constexpr int SEQLEN  = 2048;
constexpr int MROWS   = BATCH * SEQLEN;          // 4096 (GEMM M)
constexpr int MT      = MROWS / 16;              // 256 tile rows

typedef __attribute__((ext_vector_type(16))) __bf16 bf16x16;
typedef __attribute__((ext_vector_type(8)))  float  f32x8;

// ---------------------------------------------------------------------------
// NSUB 16x16 fp32 tiles (one 16 x 16*NSUB strip) of
//   C = X(MxK,row-major) * W(NxK,row-major)^T
// via v_wmma_f32_16x16x32_bf16: one A fragment reused by NSUB B fragments per
// K-step. All raw f32 for a K-step is staged into distinct float4 locals
// BEFORE conversion so the scheduler can issue all 20 global_load_b128s of a
// step back-to-back (max memory-level parallelism) instead of serializing on
// a reused staging register set.
// CDNA5 wave32 layouts:
//   A (16x32, 16-bit): lanes 0-15 -> row m=lane, K in {k0..k0+7, k0+16..k0+23};
//                      lanes 16-31 -> same rows, K shifted by +8.
//   B (32x16, 16-bit): lane n<16 -> column n, K in {k0..k0+7, k0+16..k0+23}; lanes 16-31 -> K+8.
//   C/D (16x16 f32, 8 VGPRs): VGPR r: lanes 0-15 -> (M=r, N=lane),
//                                      lanes 16-31 -> (M=8+r, N=lane-16).
// KTOT multiple of 32. If KREAL < KTOT (only the K=48 delta GEMM), a scalar
// guarded path zero-pads without reading past the W allocation.
// If SUM2, A element = X1 + X2 (used for y_fwd + y_rev before out-proj).
// ---------------------------------------------------------------------------
template <int KTOT, int KREAL, int NSUB, bool SUM2>
__device__ inline void wmma_tiles(const float* __restrict__ X1,
                                  const float* __restrict__ X2, int ldx,
                                  const float* __restrict__ W, int ldw,
                                  f32x8* acc) {
  const int lane = threadIdx.x & 31;
  const int half = lane >> 4;
  const int idx  = lane & 15;
  const float* xa1 = X1 + (size_t)idx * ldx;
  const float* xa2 = SUM2 ? (X2 + (size_t)idx * ldx) : X1;
  for (int k0 = 0; k0 < KTOT; k0 += 32) {
    const int ka = k0 + half * 8;
    const int kb = k0 + half * 8;
    bf16x16 a;
    if constexpr (KREAL == KTOT) {
      // ---- stage all loads of this K-step first (distinct SSA values) ----
      float4 av[4];
      av[0] = *(const float4*)(xa1 + ka);
      av[1] = *(const float4*)(xa1 + ka + 4);
      av[2] = *(const float4*)(xa1 + ka + 16);
      av[3] = *(const float4*)(xa1 + ka + 20);
      float4 av2[4];
      if constexpr (SUM2) {
        av2[0] = *(const float4*)(xa2 + ka);
        av2[1] = *(const float4*)(xa2 + ka + 4);
        av2[2] = *(const float4*)(xa2 + ka + 16);
        av2[3] = *(const float4*)(xa2 + ka + 20);
      }
      float4 bv[NSUB][4];
#pragma unroll
      for (int j = 0; j < NSUB; ++j) {
        const float* wb = W + (size_t)(j * 16 + idx) * ldw + kb;
#pragma unroll
        for (int q = 0; q < 4; ++q) bv[j][q] = *(const float4*)(wb + 4 * q + (q >> 1) * 8);
      }
      // ---- convert ----
      if constexpr (SUM2) {
#pragma unroll
        for (int q = 0; q < 4; ++q) {
          av[q].x += av2[q].x; av[q].y += av2[q].y;
          av[q].z += av2[q].z; av[q].w += av2[q].w;
        }
      }
#pragma unroll
      for (int q = 0; q < 4; ++q) {   // av[0..1] -> a[0..7], av[2..3] -> a[8..15]
        a[4 * q + 0] = (__bf16)av[q].x;
        a[4 * q + 1] = (__bf16)av[q].y;
        a[4 * q + 2] = (__bf16)av[q].z;
        a[4 * q + 3] = (__bf16)av[q].w;
      }
#pragma unroll
      for (int j = 0; j < NSUB; ++j) {
        bf16x16 b;
#pragma unroll
        for (int q = 0; q < 4; ++q) {
          b[4 * q + 0] = (__bf16)bv[j][q].x;
          b[4 * q + 1] = (__bf16)bv[j][q].y;
          b[4 * q + 2] = (__bf16)bv[j][q].z;
          b[4 * q + 3] = (__bf16)bv[j][q].w;
        }
        acc[j] = __builtin_amdgcn_wmma_f32_16x16x32_bf16(false, a, false, b,
                                                         (short)0, acc[j],
                                                         false, false);
      }
    } else {
      // ---- scalar guarded path (K=48 delta GEMM only) ----
#pragma unroll
      for (int i = 0; i < 8; ++i) {
        float v0 = ((ka + i)      < KREAL) ? xa1[ka + i]      : 0.f;
        float v1 = ((ka + 16 + i) < KREAL) ? xa1[ka + 16 + i] : 0.f;
        a[i]     = (__bf16)v0;
        a[i + 8] = (__bf16)v1;
      }
#pragma unroll
      for (int j = 0; j < NSUB; ++j) {
        const float* wb = W + (size_t)(j * 16 + idx) * ldw;
        bf16x16 b;
#pragma unroll
        for (int i = 0; i < 16; ++i) {
          float v = ((kb + i + (i >> 3) * 8) < KREAL) ? wb[kb + i + (i >> 3) * 8] : 0.f;
          b[i] = (__bf16)v;
        }
        acc[j] = __builtin_amdgcn_wmma_f32_16x16x32_bf16(false, a, false, b,
                                                         (short)0, acc[j],
                                                         false, false);
      }
    }
  }
}

__device__ inline float silu_f(float v) { return v / (1.f + __expf(-v)); }

// ---------------------------------------------------------------------------
// K1: xz = U @ W_in^T ; split columns into x_raw (n<1536) and z.
// 16x64 strip per wave (N=3072 -> 48 strips).
// ---------------------------------------------------------------------------
__global__ void k_gemm_in(const float* __restrict__ U,
                          const float* __restrict__ Win,
                          float* __restrict__ x_raw, float* __restrict__ z) {
  constexpr int NTG = (2 * D_INNER) / 64;  // 48
  constexpr int NT  = MT * NTG;
  const int wave = blockIdx.x * (blockDim.x >> 5) + (threadIdx.x >> 5);
  if (wave >= NT) return;
  const int tm = wave / NTG, tg = wave % NTG;
  f32x8 acc[4] = {};
  wmma_tiles<D_MODEL, D_MODEL, 4, false>(U + (size_t)tm * 16 * D_MODEL, nullptr,
                                         D_MODEL,
                                         Win + (size_t)tg * 64 * D_MODEL,
                                         D_MODEL, acc);
  const int lane = threadIdx.x & 31, half = lane >> 4, idx = lane & 15;
#pragma unroll
  for (int j = 0; j < 4; ++j) {
    const int n0 = tg * 64 + j * 16;              // 64-strip never straddles 1536
    float* dst = (n0 < D_INNER) ? x_raw : z;
    const int nloc = (n0 < D_INNER) ? n0 : n0 - D_INNER;
#pragma unroll
    for (int r = 0; r < 8; ++r) {
      const int m = tm * 16 + half * 8 + r;
      dst[(size_t)m * D_INNER + nloc + idx] = acc[j][r];
    }
  }
}

// ---------------------------------------------------------------------------
// K2: causal (fwd) + anti-causal (rev) depthwise conv(4) + bias + SiLU.
// ---------------------------------------------------------------------------
__global__ void k_conv(const float* __restrict__ x, const float* __restrict__ cw,
                       const float* __restrict__ cb,
                       float* __restrict__ xf, float* __restrict__ xr) {
  const int gid = blockIdx.x * blockDim.x + threadIdx.x;
  if (gid >= BATCH * SEQLEN * D_INNER) return;
  const int d = gid % D_INNER;
  const int t = (gid / D_INNER) % SEQLEN;
  const int b = gid / (D_INNER * SEQLEN);
  const float* xb = x + (size_t)b * SEQLEN * D_INNER + d;
  float w[D_CONV];
#pragma unroll
  for (int k = 0; k < D_CONV; ++k) w[k] = cw[d * D_CONV + k];
  const float bias = cb[d];
  float accf = bias, accr = bias;
#pragma unroll
  for (int k = 0; k < D_CONV; ++k) {
    const int tf = t - (D_CONV - 1) + k;           // causal
    if (tf >= 0) accf += w[k] * xb[(size_t)tf * D_INNER];
    const int tr = t + (D_CONV - 1) - k;           // anti-causal (flip-conv-flip)
    if (tr < SEQLEN) accr += w[k] * xb[(size_t)tr * D_INNER];
  }
  xf[gid] = silu_f(accf);
  xr[gid] = silu_f(accr);
}

// ---------------------------------------------------------------------------
// K3: x_dbl = x_dir @ W_x^T  (N=80 = dt_rank 48 | B 16 | C 16), dir = blockIdx.y.
// Single 16x16 tile per wave (N=80 is not a multiple of 64; cost ~1 GFLOP).
// ---------------------------------------------------------------------------
__global__ void k_gemm_xdbl(const float* __restrict__ xf, const float* __restrict__ xr,
                            const float* __restrict__ Wx,
                            float* __restrict__ xdf, float* __restrict__ xdr) {
  constexpr int NTN = XD / 16;  // 5
  constexpr int NT  = MT * NTN;
  const int wave = blockIdx.x * (blockDim.x >> 5) + (threadIdx.x >> 5);
  if (wave >= NT) return;
  const float* X = blockIdx.y ? xr : xf;
  float* O       = blockIdx.y ? xdr : xdf;
  const int tm = wave / NTN, tn = wave % NTN;
  f32x8 acc[1] = {};
  wmma_tiles<D_INNER, D_INNER, 1, false>(X + (size_t)tm * 16 * D_INNER, nullptr,
                                         D_INNER,
                                         Wx + (size_t)tn * 16 * D_INNER,
                                         D_INNER, acc);
  const int lane = threadIdx.x & 31, half = lane >> 4, idx = lane & 15;
#pragma unroll
  for (int r = 0; r < 8; ++r) {
    const int m = tm * 16 + half * 8 + r;
    O[(size_t)m * XD + tn * 16 + idx] = acc[0][r];
  }
}

// ---------------------------------------------------------------------------
// K4: delta = softplus(dt @ W_dt^T + b_dt), dt = x_dbl[:, :48] (ld=80), K=48->64.
// 16x64 strip per wave (scalar guarded loads; never reads past W_dt).
// ---------------------------------------------------------------------------
__global__ void k_gemm_delta(const float* __restrict__ xdf, const float* __restrict__ xdr,
                             const float* __restrict__ Wdt, const float* __restrict__ bdt,
                             float* __restrict__ df, float* __restrict__ dr) {
  constexpr int NTG = D_INNER / 64;  // 24
  constexpr int NT  = MT * NTG;
  const int wave = blockIdx.x * (blockDim.x >> 5) + (threadIdx.x >> 5);
  if (wave >= NT) return;
  const float* X = blockIdx.y ? xdr : xdf;
  float* O       = blockIdx.y ? dr : df;
  const int tm = wave / NTG, tg = wave % NTG;
  f32x8 acc[4] = {};
  wmma_tiles<64, DT_RANK, 4, false>(X + (size_t)tm * 16 * XD, nullptr, XD,
                                    Wdt + (size_t)tg * 64 * DT_RANK, DT_RANK, acc);
  const int lane = threadIdx.x & 31, half = lane >> 4, idx = lane & 15;
#pragma unroll
  for (int j = 0; j < 4; ++j) {
    const int n = tg * 64 + j * 16 + idx;
    const float bb = bdt[n];
#pragma unroll
    for (int r = 0; r < 8; ++r) {
      const int m = tm * 16 + half * 8 + r;
      const float v = acc[j][r] + bb;
      O[(size_t)m * D_INNER + n] = (v > 20.f) ? v : log1pf(__expf(v));
    }
  }
}

// ---------------------------------------------------------------------------
// K5: selective scan, one LANE per (channel d, state s): 16 lanes/channel,
// 2 channels/wave -> 3072 waves of latency hiding. Recurrence critical path
// per step is a single v_exp + FMA; the y = sum_s h*C reduction is a 4-step
// shfl_xor butterfly OFF the recurrence path (y does not feed back into h).
// Writes y_dir[t] = (sum_s h*C + u*Dp) * silu(z[t]) at original time index.
// ---------------------------------------------------------------------------
__global__ void k_scan(const float* __restrict__ df, const float* __restrict__ dr,
                       const float* __restrict__ xf, const float* __restrict__ xr,
                       const float* __restrict__ xdf, const float* __restrict__ xdr,
                       const float* __restrict__ z, const float* __restrict__ Alog,
                       const float* __restrict__ Dp,
                       float* __restrict__ yf, float* __restrict__ yr) {
  const int s    = threadIdx.x & (D_STATE - 1);
  const int dloc = threadIdx.x / D_STATE;          // 0..15 (256 threads)
  const int d    = blockIdx.x * (blockDim.x / D_STATE) + dloc;
  const int b    = blockIdx.y;
  const int dir  = blockIdx.z;
  if (d >= D_INNER) return;
  const float* delta = dir ? dr : df;
  const float* xin   = dir ? xr : xf;
  const float* xd    = dir ? xdr : xdf;
  float* yout        = dir ? yr : yf;

  const float A  = -__expf(Alog[d * D_STATE + s]);
  const float dp = Dp[d];
  float h = 0.f;

  for (int step = 0; step < SEQLEN; ++step) {
    const int t = dir ? (SEQLEN - 1 - step) : step;
    const size_t base = (size_t)b * SEQLEN + t;
    const float dl = delta[base * D_INNER + d];     // uniform within 16-group
    const float u  = xin[base * D_INNER + d];
    const float Bv = xd[base * XD + DT_RANK + s];
    const float Cv = xd[base * XD + DT_RANK + D_STATE + s];
    h = __expf(dl * A) * h + (dl * u) * Bv;
    float p = h * Cv;
    p += __shfl_xor(p, 8, 16);
    p += __shfl_xor(p, 4, 16);
    p += __shfl_xor(p, 2, 16);
    p += __shfl_xor(p, 1, 16);
    if (s == 0) {
      const float zz = z[base * D_INNER + d];
      yout[base * D_INNER + d] = (p + u * dp) * silu_f(zz);
    }
    // prefetch next step's rows into cache (global_prefetch_b8)
    const int tn2 = dir ? (t - 1) : (t + 1);
    if (tn2 >= 0 && tn2 < SEQLEN) {
      const size_t nb = (size_t)b * SEQLEN + tn2;
      __builtin_prefetch(delta + nb * D_INNER + d, 0, 3);
      __builtin_prefetch(xd + nb * XD + DT_RANK + s, 0, 3);
    }
  }
}

// ---------------------------------------------------------------------------
// K6: out = (y_f + y_r) @ W_out^T  (single GEMM replaces both directions),
// 16x64 strip per wave.
// ---------------------------------------------------------------------------
__global__ void k_gemm_out(const float* __restrict__ yf, const float* __restrict__ yr,
                           const float* __restrict__ Wout, float* __restrict__ out) {
  constexpr int NTG = D_MODEL / 64;  // 12
  constexpr int NT  = MT * NTG;
  const int wave = blockIdx.x * (blockDim.x >> 5) + (threadIdx.x >> 5);
  if (wave >= NT) return;
  const int tm = wave / NTG, tg = wave % NTG;
  f32x8 acc[4] = {};
  wmma_tiles<D_INNER, D_INNER, 4, true>(yf + (size_t)tm * 16 * D_INNER,
                                        yr + (size_t)tm * 16 * D_INNER, D_INNER,
                                        Wout + (size_t)tg * 64 * D_INNER,
                                        D_INNER, acc);
  const int lane = threadIdx.x & 31, half = lane >> 4, idx = lane & 15;
#pragma unroll
  for (int j = 0; j < 4; ++j) {
    const int n = tg * 64 + j * 16 + idx;
#pragma unroll
    for (int r = 0; r < 8; ++r) {
      const int m = tm * 16 + half * 8 + r;
      out[(size_t)m * D_MODEL + n] = acc[j][r];
    }
  }
}

// ---------------------------------------------------------------------------
extern "C" void kernel_launch(void* const* d_in, const int* in_sizes, int n_in,
                              void* d_out, int out_size, void* d_ws, size_t ws_size,
                              hipStream_t stream) {
  const float* hs   = (const float*)d_in[0];  // (B, L, 768)
  const float* Win  = (const float*)d_in[1];  // (3072, 768)
  const float* cw   = (const float*)d_in[2];  // (1536, 1, 4)
  const float* cb   = (const float*)d_in[3];  // (1536,)
  const float* Wx   = (const float*)d_in[4];  // (80, 1536)
  const float* Wdt  = (const float*)d_in[5];  // (1536, 48)
  const float* bdt  = (const float*)d_in[6];  // (1536,)
  const float* Alog = (const float*)d_in[7];  // (1536, 16)
  const float* Dp   = (const float*)d_in[8];  // (1536,)
  const float* Wout = (const float*)d_in[9];  // (768, 1536)
  float* out = (float*)d_out;

  float* ws = (float*)d_ws;
  const size_t SZ = (size_t)MROWS * D_INNER;  // 6,291,456 floats
  float* x_raw = ws;                 // pre-conv x  (reused as y_f later)
  float* yf    = ws;                 // alias: x_raw dead after K2
  float* z     = ws + 1 * SZ;
  float* xf    = ws + 2 * SZ;
  float* xr    = ws + 3 * SZ;
  float* df    = ws + 4 * SZ;
  float* dr    = ws + 5 * SZ;
  float* yr    = ws + 6 * SZ;
  float* xdf   = ws + 7 * SZ;
  float* xdr   = xdf + (size_t)MROWS * XD;

  constexpr int TPB = 256;           // 8 waves / block
  constexpr int WPB = TPB / 32;

  // K1: input projection (one GEMM for both directions)
  {
    constexpr int NT = MT * ((2 * D_INNER) / 64);
    k_gemm_in<<<(NT + WPB - 1) / WPB, TPB, 0, stream>>>(hs, Win, x_raw, z);
  }
  // K2: bidirectional depthwise conv + SiLU
  {
    const int n = BATCH * SEQLEN * D_INNER;
    k_conv<<<(n + TPB - 1) / TPB, TPB, 0, stream>>>(x_raw, cw, cb, xf, xr);
  }
  // K3: x_dbl per direction
  {
    constexpr int NT = MT * (XD / 16);
    dim3 grid((NT + WPB - 1) / WPB, 2);
    k_gemm_xdbl<<<grid, TPB, 0, stream>>>(xf, xr, Wx, xdf, xdr);
  }
  // K4: delta = softplus(dt @ W_dt^T + b) per direction
  {
    constexpr int NT = MT * (D_INNER / 64);
    dim3 grid((NT + WPB - 1) / WPB, 2);
    k_gemm_delta<<<grid, TPB, 0, stream>>>(xdf, xdr, Wdt, bdt, df, dr);
  }
  // K5: fwd + rev selective scans (grid.z = direction); 16 lanes per channel
  {
    dim3 grid(D_INNER / (TPB / D_STATE), BATCH, 2);
    k_scan<<<grid, TPB, 0, stream>>>(df, dr, xf, xr, xdf, xdr, z, Alog, Dp, yf, yr);
  }
  // K6: out = (y_f + y_r) @ W_out^T
  {
    constexpr int NT = MT * (D_MODEL / 64);
    k_gemm_out<<<(NT + WPB - 1) / WPB, TPB, 0, stream>>>(yf, yr, Wout, out);
  }
}
